// FeatureEmbedded_4518305595899
// MI455X (gfx1250) — compile-verified, matrix operand order checked by reading, not run on hardware
//
#include <hip/hip_runtime.h>
#include <hip/hip_bf16.h>
#include <stdint.h>

// Problem constants (from reference)
#define B_  32
#define W_  24
#define T_  48
#define D_  105
#define H_  512
#define N_  (B_ * W_)      // 768 sequences
#define G_  (3 * H_)       // 1536 gate width
#define DP_ 128            // D padded to multiple of 32 (WMMA K-chunk)
#define K1_ (2 * H_)       // 1024, layer-1 input width

typedef __attribute__((ext_vector_type(16))) __bf16 bf16x16;
typedef __attribute__((ext_vector_type(8)))  float  f32x8;

union FragCast { uint4 q[2]; bf16x16 v; };

// ---- bf16 <-> f32 via bit ops (no reliance on native scalar bf16 conversions) ----
__device__ __forceinline__ uint16_t f2b(float f) {
  union { float f; uint32_t u; } a; a.f = f;
  uint32_t u = a.u;
  uint32_t r = u + 0x7FFFu + ((u >> 16) & 1u);   // round-to-nearest-even
  return (uint16_t)(r >> 16);
}
__device__ __forceinline__ float b2f(uint16_t b) {
  union { uint32_t u; float f; } a; a.u = ((uint32_t)b) << 16;
  return a.f;
}

// ---- WMMA 16x16x32 bf16 fragment load ----
// ISA 16-bit A/B layout: lane L<16 -> row/col L, K = kb+{0..7} and kb+{16..23};
// lane L>=16 -> row/col L-16, K = kb+8+{0..7} and kb+24+{0..7}.
// Caller passes p = base + row*ldk + kb + ((lane&16)?8:0); we load p[0..7], p[16..23].
__device__ __forceinline__ bf16x16 load_frag(const uint16_t* p) {
  FragCast u;
  u.q[0] = *(const uint4*)(p);
  u.q[1] = *(const uint4*)(p + 16);
  return u.v;
}

__device__ __forceinline__ float sigm(float x) { return 1.0f / (1.0f + __expf(-x)); }

// =======================================================================
// Prep kernels
// =======================================================================
__global__ void idx_kernel(const int* __restrict__ lens, int* __restrict__ idxm) {
  int b = threadIdx.x;
  if (b < B_) {
    int mx = 1;
    for (int w = 0; w < W_; ++w) mx = max(mx, lens[b * W_ + w]);
    idxm[b] = mx - 1;
  }
}

// x (B,W,T,D) f32  ->  xs (T,N,DP) bf16, zero-padded D->DP, n = b*W+w
__global__ void convert_x_kernel(const float* __restrict__ x, uint16_t* __restrict__ xs) {
  int tn = blockIdx.x;          // t*N_ + n
  int t = tn / N_;
  int n = tn - t * N_;
  int dp = threadIdx.x;         // 0..127
  float v = (dp < D_) ? x[((size_t)n * T_ + t) * D_ + dp] : 0.0f;
  xs[((size_t)t * N_ + n) * DP_ + dp] = f2b(v);
}

// weight (R x C) f32 row-major -> (R x CP) bf16 zero-padded
__global__ void convert_w_kernel(const float* __restrict__ src, uint16_t* __restrict__ dst,
                                 int C, int CP) {
  int r = blockIdx.x;
  for (int c = threadIdx.x; c < CP; c += blockDim.x) {
    float v = (c < C) ? src[(size_t)r * C + c] : 0.0f;
    dst[(size_t)r * CP + c] = f2b(v);
  }
}

// =======================================================================
// Input-projection GEMM: GX (M x G_, bf16) = A (M x K, bf16) @ Wt^T + bias
// Wt is (G_ x K) row-major (out-major, K-contiguous) == B operand layout.
// Block: 128 threads = 4 waves; tile 64 rows x 64 cols; each wave 16 x 64.
// =======================================================================
__global__ __launch_bounds__(128) void gemm_gx_kernel(
    const uint16_t* __restrict__ A, const uint16_t* __restrict__ Wt,
    const float* __restrict__ bias, uint16_t* __restrict__ GX, int K) {
  const int lane = threadIdx.x & 31;
  const int wave = threadIdx.x >> 5;                 // 0..3
  const int m0   = (blockIdx.x * 4 + wave) * 16;     // row-tile base
  const int n0   = blockIdx.y * 64;                  // col-tile base
  const int half = (lane & 16) ? 8 : 0;
  const int lr   = lane & 15;

  f32x8 acc[4] = {};
  for (int kb = 0; kb < K; kb += 32) {
    bf16x16 a = load_frag(A + (size_t)(m0 + lr) * K + kb + half);
#pragma unroll
    for (int nt = 0; nt < 4; ++nt) {
      bf16x16 b = load_frag(Wt + (size_t)(n0 + nt * 16 + lr) * K + kb + half);
      acc[nt] = __builtin_amdgcn_wmma_f32_16x16x32_bf16(
          false, a, false, b, (short)0, acc[nt], false, false);
    }
  }
  // C layout: VGPR v, lanes 0-15 -> M=v, lanes 16-31 -> M=8+v; N = lane&15 (+tile)
  const int mloc = (lane & 16) ? 8 : 0;
#pragma unroll
  for (int nt = 0; nt < 4; ++nt) {
    int col = n0 + nt * 16 + lr;
    float bv = bias[col];
#pragma unroll
    for (int v = 0; v < 8; ++v) {
      int m = m0 + mloc + v;
      GX[(size_t)m * G_ + col] = f2b(acc[nt][v] + bv);
    }
  }
}

// =======================================================================
// Recurrent GRU kernel: BOTH directions in one launch (blockIdx.y = dir),
// one layer per launch. Block owns 16 sequences; h (16x512 bf16) persists
// in LDS across all T steps. Per step: gh = h @ Whh^T via 8 waves x 12
// WMMA tiles (96 tiles = G_/16), stage gh (bf16) to LDS, then 256 threads
// do gate math + hidden update.
// LDS: 16*512*2 (h) + 16*1536*2 (gh) = 64 KB dynamic.
// =======================================================================
__global__ __launch_bounds__(256) void recur_kernel(
    const uint16_t* __restrict__ gxF,   // fwd (T, N, G_) bf16 pre-activations (incl. b_ih)
    const uint16_t* __restrict__ gxR,   // bwd
    const uint16_t* __restrict__ whhF,  // fwd (G_, H_) bf16
    const uint16_t* __restrict__ whhR,  // bwd
    const float*    __restrict__ bhhF,  // fwd (G_,) f32
    const float*    __restrict__ bhhR,  // bwd
    const int*      __restrict__ lens,  // (N,)
    uint16_t*       __restrict__ h01,   // layer 0 out: (T, N, K1_) bf16
    float*          __restrict__ out,   // layer 1 out: (N, K1_) f32
    const int*      __restrict__ idxm,  // (B_,) max-len index per b
    int layer) {
  extern __shared__ char smem[];
  uint16_t* sh  = (uint16_t*)smem;                  // 16 x H_   hidden state
  uint16_t* sgh = (uint16_t*)(smem + 16 * H_ * 2);  // 16 x G_   staged gh

  const int reverse = blockIdx.y;     // 0 = fwd, 1 = bwd
  const uint16_t* gx  = reverse ? gxR  : gxF;
  const uint16_t* whh = reverse ? whhR : whhF;
  const float*    bhh = reverse ? bhhR : bhhF;

  const int tid  = threadIdx.x;
  const int lane = tid & 31;
  const int wave = tid >> 5;          // 0..7
  const int row0 = blockIdx.x * 16;   // first sequence owned by this block
  const int half = (lane & 16) ? 8 : 0;
  const int mloc = (lane & 16) ? 8 : 0;
  const int lr   = lane & 15;

  for (int p = tid; p < 16 * H_; p += 256) sh[p] = 0;
  __syncthreads();

  for (int ts = 0; ts < T_; ++ts) {
    const int t = reverse ? (T_ - 1 - ts) : ts;

    // ---- gh = h @ Whh^T : 12 output tiles per wave, K loop over H_ ----
    f32x8 acc[12] = {};
    for (int kb = 0; kb < H_; kb += 32) {
      bf16x16 a = load_frag(sh + lr * H_ + kb + half);
#pragma unroll
      for (int j = 0; j < 12; ++j) {
        int nc = (wave * 12 + j) * 16;
        bf16x16 b = load_frag(whh + (size_t)(nc + lr) * H_ + kb + half);
        acc[j] = __builtin_amdgcn_wmma_f32_16x16x32_bf16(
            false, a, false, b, (short)0, acc[j], false, false);
      }
    }
#pragma unroll
    for (int j = 0; j < 12; ++j) {
      int col = (wave * 12 + j) * 16 + lr;
#pragma unroll
      for (int v = 0; v < 8; ++v)
        sgh[(mloc + v) * G_ + col] = f2b(acc[j][v]);
    }
    __syncthreads();

    // ---- gates + hidden update; each (m,j) handled by exactly one thread ----
    for (int p = tid; p < 16 * H_; p += 256) {
      int m = p >> 9;            // row within block
      int j = p & (H_ - 1);      // hidden index
      int n = row0 + m;
      const size_t gb = ((size_t)t * N_ + n) * (size_t)G_;
      float gxr = b2f(gx[gb + j]);
      float gxz = b2f(gx[gb + H_ + j]);
      float gxn = b2f(gx[gb + 2 * H_ + j]);
      float ghr = b2f(sgh[m * G_ + j])            + bhh[j];
      float ghz = b2f(sgh[m * G_ + H_ + j])       + bhh[H_ + j];
      float ghn = b2f(sgh[m * G_ + 2 * H_ + j])   + bhh[2 * H_ + j];
      float hp  = b2f(sh[p]);
      float r  = sigm(gxr + ghr);
      float z  = sigm(gxz + ghz);
      float nn = tanhf(gxn + r * ghn);
      float hnew = (1.0f - z) * nn + z * hp;
      bool  mk = (t < lens[n]);
      sh[p] = f2b(mk ? hnew : hp);
      if (layer == 0) {
        h01[((size_t)t * N_ + n) * K1_ + (reverse ? H_ : 0) + j] = f2b(mk ? hnew : 0.0f);
      } else {
        if (!reverse) {
          if (t == idxm[n / W_]) out[(size_t)n * K1_ + j] = mk ? hnew : 0.0f;
        } else {
          if (t == 0) out[(size_t)n * K1_ + H_ + j] = mk ? hnew : 0.0f;
        }
      }
    }
    __syncthreads();
  }
}

// =======================================================================
// Host side: workspace layout + launch sequence
// =======================================================================
static constexpr size_t AU(size_t x) { return (x + 255) & ~(size_t)255; }
static constexpr size_t XS_OFF     = 0;
static constexpr size_t WIH0P_OFF  = XS_OFF     + AU((size_t)T_ * N_ * DP_ * 2);
static constexpr size_t WIH0RP_OFF = WIH0P_OFF  + AU((size_t)G_ * DP_ * 2);
static constexpr size_t WHH0_OFF   = WIH0RP_OFF + AU((size_t)G_ * DP_ * 2);
static constexpr size_t WHH0R_OFF  = WHH0_OFF   + AU((size_t)G_ * H_ * 2);
static constexpr size_t WIH1P_OFF  = WHH0R_OFF  + AU((size_t)G_ * H_ * 2);
static constexpr size_t WIH1RP_OFF = WIH1P_OFF  + AU((size_t)G_ * K1_ * 2);
static constexpr size_t WHH1_OFF   = WIH1RP_OFF + AU((size_t)G_ * K1_ * 2);
static constexpr size_t WHH1R_OFF  = WHH1_OFF   + AU((size_t)G_ * H_ * 2);
static constexpr size_t H01_OFF    = WHH1R_OFF  + AU((size_t)G_ * H_ * 2);
static constexpr size_t GXA_OFF    = H01_OFF    + AU((size_t)T_ * N_ * K1_ * 2);
static constexpr size_t GXB_OFF    = GXA_OFF    + AU((size_t)T_ * N_ * G_ * 2);
static constexpr size_t IDXM_OFF   = GXB_OFF    + AU((size_t)T_ * N_ * G_ * 2);

extern "C" void kernel_launch(void* const* d_in, const int* in_sizes, int n_in,
                              void* d_out, int out_size, void* d_ws, size_t ws_size,
                              hipStream_t stream) {
  (void)in_sizes; (void)n_in; (void)out_size; (void)ws_size;
  const float* x       = (const float*)d_in[0];
  const int*   lens    = (const int*)  d_in[1];
  const float* w_ih0   = (const float*)d_in[3];
  const float* w_hh0   = (const float*)d_in[4];
  const float* b_ih0   = (const float*)d_in[5];
  const float* b_hh0   = (const float*)d_in[6];
  const float* w_ih0r  = (const float*)d_in[7];
  const float* w_hh0r  = (const float*)d_in[8];
  const float* b_ih0r  = (const float*)d_in[9];
  const float* b_hh0r  = (const float*)d_in[10];
  const float* w_ih1   = (const float*)d_in[11];
  const float* w_hh1   = (const float*)d_in[12];
  const float* b_ih1   = (const float*)d_in[13];
  const float* b_hh1   = (const float*)d_in[14];
  const float* w_ih1r  = (const float*)d_in[15];
  const float* w_hh1r  = (const float*)d_in[16];
  const float* b_ih1r  = (const float*)d_in[17];
  const float* b_hh1r  = (const float*)d_in[18];
  float* out = (float*)d_out;

  char* ws = (char*)d_ws;
  uint16_t* xs     = (uint16_t*)(ws + XS_OFF);
  uint16_t* wih0p  = (uint16_t*)(ws + WIH0P_OFF);
  uint16_t* wih0rp = (uint16_t*)(ws + WIH0RP_OFF);
  uint16_t* whh0   = (uint16_t*)(ws + WHH0_OFF);
  uint16_t* whh0r  = (uint16_t*)(ws + WHH0R_OFF);
  uint16_t* wih1p  = (uint16_t*)(ws + WIH1P_OFF);
  uint16_t* wih1rp = (uint16_t*)(ws + WIH1RP_OFF);
  uint16_t* whh1   = (uint16_t*)(ws + WHH1_OFF);
  uint16_t* whh1r  = (uint16_t*)(ws + WHH1R_OFF);
  uint16_t* h01    = (uint16_t*)(ws + H01_OFF);
  uint16_t* gxA    = (uint16_t*)(ws + GXA_OFF);
  uint16_t* gxB    = (uint16_t*)(ws + GXB_OFF);
  int*      idxm   = (int*)     (ws + IDXM_OFF);

  // --- prep ---
  idx_kernel<<<1, 32, 0, stream>>>(lens, idxm);
  convert_x_kernel<<<T_ * N_, 128, 0, stream>>>(x, xs);
  convert_w_kernel<<<G_, 256, 0, stream>>>(w_ih0,  wih0p,  D_,  DP_);
  convert_w_kernel<<<G_, 256, 0, stream>>>(w_ih0r, wih0rp, D_,  DP_);
  convert_w_kernel<<<G_, 256, 0, stream>>>(w_hh0,  whh0,   H_,  H_);
  convert_w_kernel<<<G_, 256, 0, stream>>>(w_hh0r, whh0r,  H_,  H_);
  convert_w_kernel<<<G_, 256, 0, stream>>>(w_ih1,  wih1p,  K1_, K1_);
  convert_w_kernel<<<G_, 256, 0, stream>>>(w_ih1r, wih1rp, K1_, K1_);
  convert_w_kernel<<<G_, 256, 0, stream>>>(w_hh1,  whh1,   H_,  H_);
  convert_w_kernel<<<G_, 256, 0, stream>>>(w_hh1r, whh1r,  H_,  H_);

  const dim3 ggrid((T_ * N_) / 64, G_ / 64);         // 576 x 24
  const dim3 rgrid(N_ / 16, 2);                      // 48 blocks x 2 directions
  const size_t smem = (size_t)16 * H_ * 2 + (size_t)16 * G_ * 2;  // 64 KB

  // --- layer 0: gx = xs @ Wih^T + b_ih, then recurrence (both dirs fused) ---
  gemm_gx_kernel<<<ggrid, 128, 0, stream>>>(xs, wih0p,  b_ih0,  gxA, DP_);
  gemm_gx_kernel<<<ggrid, 128, 0, stream>>>(xs, wih0rp, b_ih0r, gxB, DP_);
  recur_kernel<<<rgrid, 256, smem, stream>>>(gxA, gxB, whh0, whh0r, b_hh0, b_hh0r,
                                             lens, h01, out, idxm, 0);

  // --- layer 1: gx1 = h01 @ Wih1^T + b_ih1 (reusing gx buffers), recurrence ---
  gemm_gx_kernel<<<ggrid, 128, 0, stream>>>(h01, wih1p,  b_ih1,  gxA, K1_);
  gemm_gx_kernel<<<ggrid, 128, 0, stream>>>(h01, wih1rp, b_ih1r, gxB, K1_);
  recur_kernel<<<rgrid, 256, smem, stream>>>(gxA, gxB, whh1, whh1r, b_hh1, b_hh1r,
                                             lens, h01, out, idxm, 1);
}
